// LiquidAudioEncoder_74947179315328
// MI455X (gfx1250) — matrix operand order repaired and checked by reading, not compiled, not run on hardware
//
#include <hip/hip_runtime.h>
#include <hip/hip_bf16.h>

// ============================================================================
// LiquidAudioEncoder on MI455X (gfx1250):
//  - conv1/conv2 as implicit-GEMM with v_wmma_f32_16x16x32_f16 (B=16 rows of
//    output time positions per tile), exact-erf GELU on f32 accumulators.
//  - CfC scan: B=16 == WMMA M dim; one 16-wave workgroup per residual block,
//    3 blocks pipelined across 3 workgroups via global acquire/release flags.
//    Weights (W*mask) pre-packed to fp16 B-fragment layout; 4.6MB -> L2-resident.
// ============================================================================

typedef __attribute__((ext_vector_type(16))) _Float16 v16h;
typedef __attribute__((ext_vector_type(8)))  _Float16 v8h;
typedef __attribute__((ext_vector_type(8)))  float    v8f;

#define BATCH   16
#define NMELS   128
#define TMEL    4096
#define DMODEL  256
#define TC1     2048
#define LSEQ    1024

// wired CfC cells: UNITS=512, MOTOR=256, CMD=102, INTER=154
#define C0_K 410
#define C0_KT 13
#define C0_N 154
#define C0_NPAD 160
#define C0_NT 10
#define C1_K 256
#define C1_KT 8
#define C1_N 102
#define C1_NPAD 112
#define C1_NT 7
#define C2_K 358
#define C2_KT 12
#define C2_N 256
#define C2_NPAD 256
#define C2_NT 16

#define TILE_HALVES 512   // one 32(K)x16(N) fp16 tile
#define CELL0_HALVES (4*C0_NT*C0_KT*TILE_HALVES)   // 266240
#define CELL1_HALVES (4*C1_NT*C1_KT*TILE_HALVES)   // 114688
#define CELL2_HALVES (4*C2_NT*C2_KT*TILE_HALVES)   // 393216
#define BLOCK_HALVES (CELL0_HALVES+CELL1_HALVES+CELL2_HALVES) // 774144
#define BIAS_PER_BLOCK (4*(C0_NPAD+C1_NPAD+C2_NPAD))          // 2112

// ---------------------------------------------------------------------------
// helpers
// ---------------------------------------------------------------------------
__device__ __forceinline__ v8f wmma_f16(v16h a, v16h b, v8f c) {
  return __builtin_amdgcn_wmma_f32_16x16x32_f16(false, a, false, b,
                                                (short)0, c, false, false);
}

// A-fragment (16x32 fp16): lane holds row M = lane&15;
// element e -> K = 32*kt + 16*(lane>>4) + e
__device__ __forceinline__ v16h load_afrag(const _Float16* zrow, int kt, int lane) {
  int kb = kt * 32 + ((lane >> 4) << 4);
  v8h lo = *(const v8h*)(zrow + kb);
  v8h hi = *(const v8h*)(zrow + kb + 8);
  v16h a;
#pragma unroll
  for (int i = 0; i < 8; ++i) { a[i] = lo[i]; a[i + 8] = hi[i]; }
  return a;
}

__device__ __forceinline__ float gelu_exact(float x) {
  return 0.5f * x * (1.f + erff(x * 0.70710678118654752f));
}

// 4 simultaneous GEMM accumulations (ff1, ff2, ta, tb) for one N-tile.
__device__ __forceinline__ void cell_mm(const _Float16* zbase, int zstride,
                                        const _Float16* wcell, int KT, int NT,
                                        int nt, int lane,
                                        v8f& a0, v8f& a1, v8f& a2, v8f& a3) {
  const int MATS = NT * KT * TILE_HALVES;
  const _Float16* zrow = zbase + (size_t)(lane & 15) * zstride;
  const _Float16* wp = wcell + (size_t)nt * KT * TILE_HALVES + lane * 16;
  for (int kt = 0; kt < KT; ++kt) {
    v16h a = load_afrag(zrow, kt, lane);
    const _Float16* wk = wp + kt * TILE_HALVES;
    v16h b0 = *(const v16h*)(wk);
    v16h b1 = *(const v16h*)(wk + MATS);
    v16h b2 = *(const v16h*)(wk + 2 * MATS);
    v16h b3 = *(const v16h*)(wk + 3 * MATS);
    a0 = wmma_f16(a, b0, a0);
    a1 = wmma_f16(a, b1, a1);
    a2 = wmma_f16(a, b2, a2);
    a3 = wmma_f16(a, b3, a3);
  }
}

// CfC activation combine; all four accs share the C layout so this is
// purely element-wise in registers. h[r] for batch row m = 8*(lane>>4)+r.
__device__ __forceinline__ void cell_act(const v8f& a0, const v8f& a1,
                                         const v8f& a2, const v8f& a3,
                                         const float* bias, int NPAD, int ncol,
                                         float* h) {
  float b0 = bias[ncol];
  float b1 = bias[NPAD + ncol];
  float b2 = bias[2 * NPAD + ncol];
  float b3 = bias[3 * NPAD + ncol];
#pragma unroll
  for (int r = 0; r < 8; ++r) {
    float f1 = tanhf(a0[r] + b0);
    float f2 = tanhf(a1[r] + b1);
    float s = (a2[r] + b2) + (a3[r] + b3);
    float ti = 1.f / (1.f + expf(-s));
    h[r] = f1 + ti * (f2 - f1);
  }
}

// ---------------------------------------------------------------------------
// weight / bias packing
// ---------------------------------------------------------------------------
struct PackWired {
  const float* w[3][4];   // [block][mat(ff1,ff2,ta,tb)] shape (K,N) row-major
  const float* mask;      // shape (K,N)
  _Float16* out;          // wired base + cell base
  int K, N, KT, NT;
};

// B-fragment layout: chunk of 512 halves per (nt,kt); element tid=L*16+e holds
// value for K = kt*32 + 16*(L>>4) + e, N = nt*16 + (L&15). Pads are zero.
__global__ __launch_bounds__(512) void pack_wired_kernel(PackWired p) {
  int wg = blockIdx.x;
  int kt = wg % p.KT; wg /= p.KT;
  int nt = wg % p.NT; wg /= p.NT;
  int mat = wg % 4;
  int blk = wg / 4;
  int tid = threadIdx.x;
  int L = tid >> 4, e = tid & 15;
  int k = kt * 32 + ((L >> 4) << 4) + e;
  int n = nt * 16 + (L & 15);
  float v = 0.f;
  if (k < p.K && n < p.N) {
    size_t o = (size_t)k * p.N + n;
    v = p.w[blk][mat][o] * p.mask[o];
  }
  size_t off = (size_t)blk * BLOCK_HALVES +
               (size_t)mat * ((size_t)p.NT * p.KT * TILE_HALVES) +
               ((size_t)nt * p.KT + kt) * TILE_HALVES + tid;
  p.out[off] = (_Float16)v;
}

// conv weights (CO=256, CI, 3) -> B layout with K = kk*CI + ci
__global__ __launch_bounds__(512) void pack_conv_kernel(const float* w, _Float16* out,
                                                        int CI, int KT) {
  int wg = blockIdx.x;
  int kt = wg % KT;
  int nt = wg / KT;
  int tid = threadIdx.x;
  int L = tid >> 4, e = tid & 15;
  int k = kt * 32 + ((L >> 4) << 4) + e;
  int n = nt * 16 + (L & 15);
  int ci = k % CI;
  int kk = k / CI;
  float v = (kk < 3) ? w[((size_t)n * CI + ci) * 3 + kk] : 0.f;
  out[((size_t)nt * KT + kt) * TILE_HALVES + tid] = (_Float16)v;
}

struct PackBias {
  const float* b[3][3][4];  // [block][cell][mat]
  float* out;               // [block][2112]
};

__global__ void pack_bias_kernel(PackBias p) {
  int i = blockIdx.x * blockDim.x + threadIdx.x;
  if (i >= 3 * BIAS_PER_BLOCK) return;
  int blk = i / BIAS_PER_BLOCK;
  int loc = i % BIAS_PER_BLOCK;
  int cell, mat, u, U;
  if (loc < 4 * C0_NPAD)       { cell = 0; U = C0_N; mat = loc / C0_NPAD; u = loc % C0_NPAD; }
  else if (loc < 4 * C0_NPAD + 4 * C1_NPAD) {
    int l = loc - 4 * C0_NPAD;  cell = 1; U = C1_N; mat = l / C1_NPAD; u = l % C1_NPAD;
  } else {
    int l = loc - 4 * C0_NPAD - 4 * C1_NPAD; cell = 2; U = C2_N; mat = l / C2_NPAD; u = l % C2_NPAD;
  }
  p.out[i] = (u < U) ? p.b[blk][cell][mat][u] : 0.f;
}

__global__ void init_flags_kernel(int* f) {
  if (threadIdx.x < 16) f[threadIdx.x] = 0;
}

// ---------------------------------------------------------------------------
// conv1d (stride 2, pad 1) + GELU via implicit-GEMM WMMA
// ---------------------------------------------------------------------------
__global__ __launch_bounds__(256) void conv1_kernel(const float* __restrict__ mel,
                                                    const float* __restrict__ bias,
                                                    const _Float16* __restrict__ wpack,
                                                    float* __restrict__ out) {
  int b = blockIdx.x, ttile = blockIdx.y, ngrp = blockIdx.z;
  __shared__ __align__(32) _Float16 patch[16][384];
  int tid = threadIdx.x, lane = tid & 31, wave = tid >> 5;
  for (int i = tid; i < 16 * 384; i += 256) {
    int row = i / 384, k = i % 384;
    int ci = k & 127, kk = k >> 7;
    int tg = (ttile * 16 + row) * 2 + kk - 1;
    float v = (tg >= 0 && tg < TMEL) ? mel[((size_t)b * NMELS + ci) * TMEL + tg] : 0.f;
    patch[row][k] = (_Float16)v;
  }
  __syncthreads();
  int nt = ngrp * 8 + wave;
  v8f acc = {};
  const _Float16* zrow = &patch[lane & 15][0];
  const _Float16* wp = wpack + (size_t)nt * 12 * TILE_HALVES + lane * 16;
  for (int kt = 0; kt < 12; ++kt) {
    v16h a = load_afrag(zrow, kt, lane);
    v16h bm = *(const v16h*)(wp + kt * TILE_HALVES);
    acc = wmma_f16(a, bm, acc);
  }
  int co = nt * 16 + (lane & 15);
  float bi = bias[co];
#pragma unroll
  for (int r = 0; r < 8; ++r) {
    int tp = ttile * 16 + ((lane >> 4) << 3) + r;
    out[((size_t)b * DMODEL + co) * TC1 + tp] = gelu_exact(acc[r] + bi);
  }
}

__global__ __launch_bounds__(256) void conv2_kernel(const float* __restrict__ xin,
                                                    const float* __restrict__ bias,
                                                    const _Float16* __restrict__ wpack,
                                                    float* __restrict__ out) {
  int b = blockIdx.x, ttile = blockIdx.y, ngrp = blockIdx.z;
  __shared__ __align__(32) _Float16 patch[16][768];
  int tid = threadIdx.x, lane = tid & 31, wave = tid >> 5;
  for (int i = tid; i < 16 * 768; i += 256) {
    int row = i / 768, k = i % 768;
    int ci = k & 255, kk = k >> 8;
    int tg = (ttile * 16 + row) * 2 + kk - 1;
    float v = (tg >= 0 && tg < TC1) ? xin[((size_t)b * DMODEL + ci) * TC1 + tg] : 0.f;
    patch[row][k] = (_Float16)v;
  }
  __syncthreads();
  int nt = ngrp * 8 + wave;
  v8f acc = {};
  const _Float16* zrow = &patch[lane & 15][0];
  const _Float16* wp = wpack + (size_t)nt * 24 * TILE_HALVES + lane * 16;
  for (int kt = 0; kt < 24; ++kt) {
    v16h a = load_afrag(zrow, kt, lane);
    v16h bm = *(const v16h*)(wp + kt * TILE_HALVES);
    acc = wmma_f16(a, bm, acc);
  }
  int co = nt * 16 + (lane & 15);
  float bi = bias[co];
#pragma unroll
  for (int r = 0; r < 8; ++r) {
    int tp = ttile * 16 + ((lane >> 4) << 3) + r;
    // output layout (B, L, D)
    out[((size_t)b * LSEQ + tp) * DMODEL + co] = gelu_exact(acc[r] + bi);
  }
}

// ---------------------------------------------------------------------------
// CfC scan: 3 workgroups, one residual block each, pipelined via flags.
// ---------------------------------------------------------------------------
struct ScanArgs {
  const float* ln_scale[3];
  const float* ln_bias[3];
  const _Float16* wired;
  const float* biasp;
  const float* x0;
  float* x1;
  float* x2;
  float* xfinal;
  int* flags;
};

__global__ __launch_bounds__(512) void scan_kernel(ScanArgs p) {
  const int blk = blockIdx.x;          // residual block id 0..2
  const int tid = threadIdx.x;
  const int lane = tid & 31, wave = tid >> 5;

  // z buffers (fp16): z0 = [y(256) | h0(154) pad->416], z1 = [h0(154)|h1(102)],
  // z2 = [h1(102) | h2(256) pad->384]. xb holds x_t for residual.
  __shared__ __align__(32) _Float16 z0[16][416];
  __shared__ __align__(32) _Float16 z1[16][256];
  __shared__ __align__(32) _Float16 z2[16][384];
  __shared__ float xb[16][256];

  for (int i = tid; i < 16 * 160; i += 512) z0[i / 160][256 + i % 160] = (_Float16)0.f;
  for (int i = tid; i < 16 * 256; i += 512) z1[i >> 8][i & 255] = (_Float16)0.f;
  for (int i = tid; i < 16 * 384; i += 512) z2[i / 384][i % 384] = (_Float16)0.f;

  const float* xin;
  float* xout;
  if (blk == 0)      { xin = p.x0; xout = p.x1; }
  else if (blk == 1) { xin = p.x1; xout = p.x2; }
  else               { xin = p.x2; xout = p.xfinal; }

  const _Float16* wblk = p.wired + (size_t)blk * BLOCK_HALVES;
  const float* bblk = p.biasp + blk * BIAS_PER_BLOCK;
  const float* lns = p.ln_scale[blk];
  const float* lnb = p.ln_bias[blk];
  int* flg = p.flags;
  __syncthreads();

  for (int t = 0; t < LSEQ; ++t) {
    // --- wait until producer block published x_in[:, t, :] ---
    if (blk > 0) {
      if (tid == 0) {
        while (__hip_atomic_load(&flg[blk - 1], __ATOMIC_ACQUIRE,
                                 __HIP_MEMORY_SCOPE_AGENT) < t + 1) {
          __builtin_amdgcn_s_sleep(1);
        }
      }
      __syncthreads();
    }

    // --- stage x_t (16x256 f32) ---
    for (int i = tid; i < 16 * 256; i += 512) {
      int m = i >> 8, k = i & 255;
      xb[m][k] = xin[((size_t)m * LSEQ + t) * DMODEL + k];
    }
    __syncthreads();

    // --- layernorm: wave w owns batch row w ---
    {
      int row = wave;
      float v[8], s = 0.f, s2 = 0.f;
#pragma unroll
      for (int i = 0; i < 8; ++i) {
        v[i] = xb[row][lane * 8 + i];
        s += v[i];
        s2 += v[i] * v[i];
      }
#pragma unroll
      for (int off = 16; off > 0; off >>= 1) {
        s  += __shfl_xor(s, off, 32);
        s2 += __shfl_xor(s2, off, 32);
      }
      float mu = s * (1.f / 256.f);
      float var = s2 * (1.f / 256.f) - mu * mu;
      float inv = rsqrtf(var + 1e-5f);
#pragma unroll
      for (int i = 0; i < 8; ++i) {
        int k = lane * 8 + i;
        float y = (v[i] - mu) * inv * lns[k] + lnb[k];
        z0[row][k] = (_Float16)y;
      }
    }
    __syncthreads();

    // --- cell 0: z=[y,h0] K=416, N-tiles=10 ---
    if (wave < C0_NT) {
      v8f a0 = {}, a1 = {}, a2 = {}, a3 = {};
      cell_mm(&z0[0][0], 416, wblk, C0_KT, C0_NT, wave, lane, a0, a1, a2, a3);
      float h[8];
      int ncol = wave * 16 + (lane & 15);
      cell_act(a0, a1, a2, a3, bblk, C0_NPAD, ncol, h);
#pragma unroll
      for (int r = 0; r < 8; ++r) {
        int m = ((lane >> 4) << 3) + r;
        z0[m][256 + ncol] = (_Float16)h[r];           // pads write exact 0
        if (ncol < C0_N) z1[m][ncol] = (_Float16)h[r];
      }
    }
    __syncthreads();

    // --- cell 1: z=[h0,h1] K=256, N-tiles=7 ---
    if (wave < C1_NT) {
      v8f a0 = {}, a1 = {}, a2 = {}, a3 = {};
      cell_mm(&z1[0][0], 256, wblk + CELL0_HALVES, C1_KT, C1_NT, wave, lane,
              a0, a1, a2, a3);
      float h[8];
      int ncol = wave * 16 + (lane & 15);
      cell_act(a0, a1, a2, a3, bblk + 4 * C0_NPAD, C1_NPAD, ncol, h);
#pragma unroll
      for (int r = 0; r < 8; ++r) {
        int m = ((lane >> 4) << 3) + r;
        if (ncol < C1_N) {
          z1[m][C0_N + ncol] = (_Float16)h[r];
          z2[m][ncol]        = (_Float16)h[r];
        }
      }
    }
    __syncthreads();

    // --- cell 2 (motor): z=[h1,h2] K=384, N-tiles=16; residual + output ---
    {
      v8f a0 = {}, a1 = {}, a2 = {}, a3 = {};
      cell_mm(&z2[0][0], 384, wblk + CELL0_HALVES + CELL1_HALVES, C2_KT, C2_NT,
              wave, lane, a0, a1, a2, a3);
      float h[8];
      int ncol = wave * 16 + (lane & 15);
      cell_act(a0, a1, a2, a3, bblk + 4 * C0_NPAD + 4 * C1_NPAD, C2_NPAD, ncol, h);
#pragma unroll
      for (int r = 0; r < 8; ++r) {
        int m = ((lane >> 4) << 3) + r;
        z2[m][C1_N + ncol] = (_Float16)h[r];
        xout[((size_t)m * LSEQ + t) * DMODEL + ncol] = xb[m][ncol] + h[r];
      }
    }
    __threadfence();
    __syncthreads();
    if (tid == 0)
      __hip_atomic_store(&flg[blk], t + 1, __ATOMIC_RELEASE,
                         __HIP_MEMORY_SCOPE_AGENT);
  }
}

// ---------------------------------------------------------------------------
// host
// ---------------------------------------------------------------------------
extern "C" void kernel_launch(void* const* d_in, const int* in_sizes, int n_in,
                              void* d_out, int out_size, void* d_ws, size_t ws_size,
                              hipStream_t stream) {
  (void)in_sizes; (void)n_in; (void)out_size; (void)ws_size;

  // assumed flattening: setup_inputs() dict insertion order
  const float* mel = (const float*)d_in[0];
  const float* c1w = (const float*)d_in[1];
  const float* c1b = (const float*)d_in[2];
  const float* c2w = (const float*)d_in[3];
  const float* c2b = (const float*)d_in[4];
  const float* lns[3]; const float* lnbv[3];
  const float* W[3][3][4]; const float* Bv[3][3][4];
  int idx = 5;
  for (int b = 0; b < 3; ++b) {
    lns[b] = (const float*)d_in[idx++];
    lnbv[b] = (const float*)d_in[idx++];
    for (int c = 0; c < 3; ++c)
      for (int m = 0; m < 4; ++m) {
        W[b][c][m]  = (const float*)d_in[idx++];
        Bv[b][c][m] = (const float*)d_in[idx++];
      }
  }
  const float* mask[3];
  for (int c = 0; c < 3; ++c) mask[c] = (const float*)d_in[idx++];

  // workspace carve
  char* ws = (char*)d_ws;
  size_t off = 0;
  auto carve = [&](size_t bytes) -> char* {
    char* p = ws + off;
    off += (bytes + 255) & ~(size_t)255;
    return p;
  };
  int*       flags = (int*)carve(64);
  float*     biasp = (float*)carve((size_t)3 * BIAS_PER_BLOCK * 4);
  _Float16*  c1p   = (_Float16*)carve((size_t)16 * 12 * TILE_HALVES * 2);
  _Float16*  c2p   = (_Float16*)carve((size_t)16 * 24 * TILE_HALVES * 2);
  _Float16*  wired = (_Float16*)carve((size_t)3 * BLOCK_HALVES * 2);
  float*     c1out = (float*)carve((size_t)BATCH * DMODEL * TC1 * 4);
  float*     x0    = (float*)carve((size_t)BATCH * LSEQ * DMODEL * 4);
  float*     x1    = (float*)carve((size_t)BATCH * LSEQ * DMODEL * 4);
  float*     x2    = (float*)carve((size_t)BATCH * LSEQ * DMODEL * 4);

  init_flags_kernel<<<1, 64, 0, stream>>>(flags);

  PackBias pb;
  for (int b = 0; b < 3; ++b)
    for (int c = 0; c < 3; ++c)
      for (int m = 0; m < 4; ++m) pb.b[b][c][m] = Bv[b][c][m];
  pb.out = biasp;
  pack_bias_kernel<<<(3 * BIAS_PER_BLOCK + 255) / 256, 256, 0, stream>>>(pb);

  pack_conv_kernel<<<16 * 12, 512, 0, stream>>>(c1w, c1p, 128, 12);
  pack_conv_kernel<<<16 * 24, 512, 0, stream>>>(c2w, c2p, 256, 24);

  {
    PackWired pw; pw.mask = mask[0]; pw.out = wired;
    pw.K = C0_K; pw.N = C0_N; pw.KT = C0_KT; pw.NT = C0_NT;
    for (int b = 0; b < 3; ++b) for (int m = 0; m < 4; ++m) pw.w[b][m] = W[b][0][m];
    pack_wired_kernel<<<3 * 4 * C0_NT * C0_KT, 512, 0, stream>>>(pw);
  }
  {
    PackWired pw; pw.mask = mask[1]; pw.out = wired + CELL0_HALVES;
    pw.K = C1_K; pw.N = C1_N; pw.KT = C1_KT; pw.NT = C1_NT;
    for (int b = 0; b < 3; ++b) for (int m = 0; m < 4; ++m) pw.w[b][m] = W[b][1][m];
    pack_wired_kernel<<<3 * 4 * C1_NT * C1_KT, 512, 0, stream>>>(pw);
  }
  {
    PackWired pw; pw.mask = mask[2]; pw.out = wired + CELL0_HALVES + CELL1_HALVES;
    pw.K = C2_K; pw.N = C2_N; pw.KT = C2_KT; pw.NT = C2_NT;
    for (int b = 0; b < 3; ++b) for (int m = 0; m < 4; ++m) pw.w[b][m] = W[b][2][m];
    pack_wired_kernel<<<3 * 4 * C2_NT * C2_KT, 512, 0, stream>>>(pw);
  }

  conv1_kernel<<<dim3(16, 128, 2), 256, 0, stream>>>(mel, c1b, c1p, c1out);
  conv2_kernel<<<dim3(16, 64, 2), 256, 0, stream>>>(c1out, c2b, c2p, x0);

  ScanArgs sa;
  for (int b = 0; b < 3; ++b) { sa.ln_scale[b] = lns[b]; sa.ln_bias[b] = lnbv[b]; }
  sa.wired = wired; sa.biasp = biasp;
  sa.x0 = x0; sa.x1 = x1; sa.x2 = x2; sa.xfinal = (float*)d_out;
  sa.flags = flags;
  scan_kernel<<<3, 512, 0, stream>>>(sa);
}
